// ERPTransformerLayer_68822555951754
// MI455X (gfx1250) — matrix operand order repaired and checked by reading, not compiled, not run on hardware
//
#include <hip/hip_runtime.h>
#include <hip/hip_bf16.h>

// ---------------------------------------------------------------------------
// ERP graph-transformer layer for gfx1250 (MI455X).
// BS=4, N=256, DX=256, DE=64, NH=8, DF=32, FFX=1024, FFE=128.
// Strategy: bf16 WMMA (v_wmma_f32_16x16x32_bf16, f32 accumulate) for all
// GEMMs; fused per-(b,i) edge kernel holding Y (256x256 f32) in LDS.
// ---------------------------------------------------------------------------

typedef __bf16 bf16;
typedef __attribute__((ext_vector_type(16))) __bf16 v16bf;
typedef __attribute__((ext_vector_type(8)))  float  v8f;

#define WMMA_BF16(a, b, c) \
  __builtin_amdgcn_wmma_f32_16x16x32_bf16(false, (a), false, (b), (short)0, (c), false, false)

// ---- fragment loaders -----------------------------------------------------
// A (16x32 bf16, M x K): lane m = lane&15; element e -> K = (e<8?e:e+8) + 8*(lane>>4)
__device__ __forceinline__ v16bf load_a_bf16(const bf16* A, int lda, int m0, int k0, int lane) {
  const bf16* p = A + (size_t)(m0 + (lane & 15)) * lda + k0 + ((lane >> 4) << 3);
  v16bf a;
#pragma unroll
  for (int e = 0; e < 16; ++e) a[e] = p[e < 8 ? e : e + 8];
  return a;
}

__device__ __forceinline__ v16bf load_a_f32(const float* A, int lda, int m0, int k0, int lane) {
  const float* p = A + (size_t)(m0 + (lane & 15)) * lda + k0 + ((lane >> 4) << 3);
  v16bf a;
#pragma unroll
  for (int e = 0; e < 16; ++e) a[e] = (bf16)p[e < 8 ? e : e + 8];
  return a;
}

// B fragments pre-swizzled: Bp[((kt*ntiles + nt)*32 + lane)*16 + e]
__device__ __forceinline__ v16bf load_b_packed(const bf16* Bp, int ntiles, int kt, int nt, int lane) {
  return *(const v16bf*)(Bp + (((size_t)(kt * ntiles + nt)) * 32 + lane) * 16);
}

// ---- prep kernels ---------------------------------------------------------
__global__ void f32_to_bf16_kernel(const float* __restrict__ in, bf16* __restrict__ out, int n) {
  int i = blockIdx.x * blockDim.x + threadIdx.x;
  if (i < n) out[i] = (bf16)in[i];
}

// Pack row-major W[K x N] (f32) into WMMA B-fragment order (bf16).
// B element e of lane at (kt,nt) <-> W[kt*32 + (e<8?e:e+8) + 8*(lane>>4)][nt*16 + (lane&15)]
__global__ void pack_b_kernel(const float* __restrict__ W, bf16* __restrict__ out, int K, int Ncols) {
  int idx = blockIdx.x * blockDim.x + threadIdx.x;
  if (idx >= K * Ncols) return;
  int k = idx / Ncols, n = idx % Ncols;
  int kt = k >> 5, kr = k & 31;
  int nt = n >> 4, nl = n & 15;
  int lane = (((kr >> 3) & 1) << 4) | nl;
  int e = (kr & 7) + ((kr >> 4) << 3);
  int ntiles = Ncols >> 4;
  out[(((size_t)(kt * ntiles + nt)) * 32 + lane) * 16 + e] = (bf16)W[idx];
}

// ---- generic bf16 WMMA GEMM: C = A @ B + bias, optional relu / row-mask ---
__global__ void __launch_bounds__(256)
gemm_bf16_kernel(const bf16* __restrict__ A, const bf16* __restrict__ Bp,
                 const float* __restrict__ bias, const int* __restrict__ rowmask,
                 float* __restrict__ Cf, bf16* __restrict__ Cb,
                 int M, int Ncols, int K, int relu) {
  int lane = threadIdx.x & 31;
  int gw   = (blockIdx.x * blockDim.x + threadIdx.x) >> 5;
  int gws  = (gridDim.x * blockDim.x) >> 5;
  int nt = Ncols >> 4;
  int tiles = (M >> 4) * nt;
  int ktn = K >> 5;
  for (int t = gw; t < tiles; t += gws) {         // wave-uniform loop (EXEC all 1s)
    int mtile = t / nt, ntile = t % nt;
    v8f c = {};
    for (int kt = 0; kt < ktn; ++kt) {
      v16bf a = load_a_bf16(A, K, mtile * 16, kt * 32, lane);
      v16bf b = load_b_packed(Bp, nt, kt, ntile, lane);
      c = WMMA_BF16(a, b, c);
    }
    int col = ntile * 16 + (lane & 15);
    int rbase = mtile * 16 + ((lane >> 4) << 3);
    float bv = bias ? bias[col] : 0.0f;
#pragma unroll
    for (int v = 0; v < 8; ++v) {
      int row = rbase + v;
      float val = c[v] + bv;
      if (relu) val = fmaxf(val, 0.0f);
      if (rowmask) val *= (float)rowmask[row];
      if (Cf) Cf[(size_t)row * Ncols + col] = val;
      if (Cb) Cb[(size_t)row * Ncols + col] = (bf16)val;
    }
  }
}

// ---- LayerNorm(a + r) * g + beta ; one thread per row ---------------------
__global__ void ln_add_kernel(const float* __restrict__ a, const float* __restrict__ r,
                              const float* __restrict__ g, const float* __restrict__ be,
                              float* __restrict__ outf, bf16* __restrict__ outb,
                              int rows, int D) {
  int row = blockIdx.x * blockDim.x + threadIdx.x;
  if (row >= rows) return;
  const float* pa = a + (size_t)row * D;
  const float* pr = r + (size_t)row * D;
  float s = 0.f, s2 = 0.f;
  for (int f = 0; f < D; ++f) { float v = pa[f] + pr[f]; s += v; s2 += v * v; }
  float mean = s / (float)D;
  float var  = s2 / (float)D - mean * mean;
  float rs   = rsqrtf(var + 1e-5f);
  for (int f = 0; f < D; ++f) {
    float v = pa[f] + pr[f];
    float o = (v - mean) * rs * g[f] + be[f];
    if (outf) outf[(size_t)row * D + f] = o;
    if (outb) outb[(size_t)row * D + f] = (bf16)o;
  }
}

// ---- fused edge kernel: one workgroup per (b, i) --------------------------
// LDS: [0,256KB) Y f32 ; later re-partitioned as newE/E1n/H/ffE.
//      [256KB,288KB) E-row bf16 (later E1n bf16) ; [288KB,+1KB) node mask.
__global__ void __launch_bounds__(256)
edge_kernel(const float* __restrict__ E, const int* __restrict__ node_mask,
            const float* __restrict__ Qf, const float* __restrict__ Kf, const float* __restrict__ Vf,
            const bf16* __restrict__ Wem_p, const bf16* __restrict__ Wea_p, const bf16* __restrict__ Weo_p,
            const bf16* __restrict__ WE1_p, const bf16* __restrict__ WE2_p,
            const float* __restrict__ bem, const float* __restrict__ bea, const float* __restrict__ beo,
            const float* __restrict__ bE1, const float* __restrict__ bE2,
            const float* __restrict__ gE1, const float* __restrict__ bnE1,
            const float* __restrict__ gE2, const float* __restrict__ bnE2,
            float* __restrict__ WV, float* __restrict__ Eout) {
  extern __shared__ char smem[];
  float* Ybuf = (float*)smem;                      // 256*256 f32 = 256 KB
  bf16*  Abf  = (bf16*)(smem + 262144);            // 256*64 bf16 = 32 KB
  float* xms  = (float*)(smem + 262144 + 32768);   // 256 f32
  float* nE   = (float*)smem;                      // alias: 256*64 f32
  float* E1nf = (float*)(smem + 65536);            // alias: 256*64 f32
  bf16*  Hb   = (bf16*)(smem + 131072);            // alias: 256*128 bf16
  float* ffE  = (float*)(smem + 196608);           // alias: 256*64 f32

  const int tid = threadIdx.x;
  const int lane = tid & 31;
  const int wave = tid >> 5;
  const int b = blockIdx.x >> 8;
  const int i = blockIdx.x & 255;
  const float scale = 0.17677669529663687f;        // 1/sqrt(DF=32)

  // phase 0: stage E[b,i,:,:] as bf16; stage mask row
  const float* Ei = E + (size_t)(b * 256 + i) * (256 * 64);
  for (int idx = tid; idx < 256 * 64; idx += 256) Abf[idx] = (bf16)Ei[idx];
  xms[tid] = (float)node_mask[b * 256 + tid];
  __syncthreads();
  const float xmi = xms[i];

  // phase 1: E1 = (E@Wem+bem)*ep, E2 = (E@Wea+bea)*ep, Y = Q*K*scale*(E1+1)+E2
  const float* Qrow = Qf + (size_t)(b * 256 + i) * 256;
  for (int t = wave; t < 256; t += 8) {
    int mtile = t >> 4, ntile = t & 15;
    v8f c1 = {}, c2 = {};
#pragma unroll
    for (int kt = 0; kt < 2; ++kt) {
      v16bf a = load_a_bf16(Abf, 64, mtile * 16, kt * 32, lane);
      c1 = WMMA_BF16(a, load_b_packed(Wem_p, 16, kt, ntile, lane), c1);
      c2 = WMMA_BF16(a, load_b_packed(Wea_p, 16, kt, ntile, lane), c2);
    }
    int f = ntile * 16 + (lane & 15);
    int jbase = mtile * 16 + ((lane >> 4) << 3);
    float qv = Qrow[f] * scale;
    float bm = bem[f], ba = bea[f];
#pragma unroll
    for (int v = 0; v < 8; ++v) {
      int j = jbase + v;
      float ep = xmi * xms[j];
      float e1 = (c1[v] + bm) * ep;
      float e2 = (c2[v] + ba) * ep;
      float kv = Kf[(size_t)(b * 256 + j) * 256 + f];
      Ybuf[j * 256 + f] = qv * kv * (e1 + 1.0f) + e2;
    }
  }
  __syncthreads();

  // phase 2a: newE fragments = Y @ Weo (M=256,N=64,K=256), kept in registers
  v8f accN[8];
  {
    int idx = 0;
    for (int t = wave; t < 64; t += 8, ++idx) {
      int mtile = t >> 2, ntile = t & 3;
      v8f c = {};
#pragma unroll
      for (int kt = 0; kt < 8; ++kt) {
        v16bf a = load_a_f32(Ybuf, 256, mtile * 16, kt * 32, lane);
        c = WMMA_BF16(a, load_b_packed(Weo_p, 4, kt, ntile, lane), c);
      }
      accN[idx] = c;
    }
  }

  // phase 2b: per-feature online softmax over j + attn·V reduction
  {
    float m = -3.0e38f, l = 0.f, acc = 0.f;
    for (int j = 0; j < 256; ++j) {
      if (xms[j] > 0.f) {                 // uniform across threads
        float y = Ybuf[j * 256 + tid];
        float nm = fmaxf(m, y);
        float corr = __expf(m - nm);
        float p = __expf(y - nm);
        l = l * corr + p;
        acc = acc * corr + p * Vf[(size_t)(b * 256 + j) * 256 + tid];
        m = nm;
      }
    }
    WV[(size_t)(b * 256 + i) * 256 + tid] = (l > 0.f) ? (acc / l) : 0.f;
  }
  __syncthreads();   // all readers of Ybuf done

  // phase 3: newE -> LDS (overwrites Ybuf region), with bias + epair mask
  {
    int idx = 0;
    for (int t = wave; t < 64; t += 8, ++idx) {
      int mtile = t >> 2, ntile = t & 3;
      int f = ntile * 16 + (lane & 15);
      int jbase = mtile * 16 + ((lane >> 4) << 3);
      float bo = beo[f];
#pragma unroll
      for (int v = 0; v < 8; ++v) {
        int j = jbase + v;
        nE[j * 64 + f] = (accN[idx][v] + bo) * (xmi * xms[j]);
      }
    }
  }
  __syncthreads();

  // phase 4: E1n = LN(E + newE); thread = row j
  {
    const float* er = Ei + (size_t)tid * 64;
    const float* nr = nE + tid * 64;
    float s = 0.f, s2 = 0.f;
    for (int f = 0; f < 64; ++f) { float v = er[f] + nr[f]; s += v; s2 += v * v; }
    float mean = s * (1.f / 64.f);
    float var  = s2 * (1.f / 64.f) - mean * mean;
    float rs   = rsqrtf(var + 1e-5f);
    for (int f = 0; f < 64; ++f) {
      float v = er[f] + nr[f];
      float o = (v - mean) * rs * gE1[f] + bnE1[f];
      E1nf[tid * 64 + f] = o;
      Abf[tid * 64 + f]  = (bf16)o;       // A-operand for FFN
    }
  }
  __syncthreads();

  // phase 5: H = relu(E1n @ WE1 + bE1)  (M=256,N=128,K=64)
  for (int t = wave; t < 128; t += 8) {
    int mtile = t >> 3, ntile = t & 7;
    v8f c = {};
#pragma unroll
    for (int kt = 0; kt < 2; ++kt) {
      v16bf a = load_a_bf16(Abf, 64, mtile * 16, kt * 32, lane);
      c = WMMA_BF16(a, load_b_packed(WE1_p, 8, kt, ntile, lane), c);
    }
    int col = ntile * 16 + (lane & 15);
    int jbase = mtile * 16 + ((lane >> 4) << 3);
    float bv = bE1[col];
#pragma unroll
    for (int v = 0; v < 8; ++v)
      Hb[(jbase + v) * 128 + col] = (bf16)fmaxf(c[v] + bv, 0.f);
  }
  __syncthreads();

  // phase 6: ffE = H @ WE2 + bE2  (M=256,N=64,K=128)
  for (int t = wave; t < 64; t += 8) {
    int mtile = t >> 2, ntile = t & 3;
    v8f c = {};
#pragma unroll
    for (int kt = 0; kt < 4; ++kt) {
      v16bf a = load_a_bf16(Hb, 128, mtile * 16, kt * 32, lane);
      c = WMMA_BF16(a, load_b_packed(WE2_p, 4, kt, ntile, lane), c);
    }
    int col = ntile * 16 + (lane & 15);
    int jbase = mtile * 16 + ((lane >> 4) << 3);
    float bv = bE2[col];
#pragma unroll
    for (int v = 0; v < 8; ++v)
      ffE[(jbase + v) * 64 + col] = c[v] + bv;
  }
  __syncthreads();

  // phase 7: Eout = LN(E1n + ffE); thread = row j
  {
    const float* ar = E1nf + tid * 64;
    const float* fr = ffE + tid * 64;
    float s = 0.f, s2 = 0.f;
    for (int f = 0; f < 64; ++f) { float v = ar[f] + fr[f]; s += v; s2 += v * v; }
    float mean = s * (1.f / 64.f);
    float var  = s2 * (1.f / 64.f) - mean * mean;
    float rs   = rsqrtf(var + 1e-5f);
    float* outr = Eout + (size_t)((b * 256 + i) * 256 + tid) * 64;
    for (int f = 0; f < 64; ++f) {
      float v = ar[f] + fr[f];
      outr[f] = (v - mean) * rs * gE2[f] + bnE2[f];
    }
  }
}

// ---------------------------------------------------------------------------
extern "C" void kernel_launch(void* const* d_in, const int* in_sizes, int n_in,
                              void* d_out, int out_size, void* d_ws, size_t ws_size,
                              hipStream_t stream) {
  (void)in_sizes; (void)n_in; (void)out_size; (void)ws_size;
  const float* X   = (const float*)d_in[0];
  const float* E   = (const float*)d_in[1];
  const int*   nm  = (const int*)d_in[2];
  const float* Wq  = (const float*)d_in[3];  const float* bq   = (const float*)d_in[4];
  const float* Wk  = (const float*)d_in[5];  const float* bk   = (const float*)d_in[6];
  const float* Wv  = (const float*)d_in[7];  const float* bv   = (const float*)d_in[8];
  const float* Wem = (const float*)d_in[9];  const float* bem  = (const float*)d_in[10];
  const float* Wea = (const float*)d_in[11]; const float* bea  = (const float*)d_in[12];
  const float* Wxo = (const float*)d_in[13]; const float* bxo  = (const float*)d_in[14];
  const float* Weo = (const float*)d_in[15]; const float* beo  = (const float*)d_in[16];
  const float* WX1 = (const float*)d_in[17]; const float* bX1  = (const float*)d_in[18];
  const float* WX2 = (const float*)d_in[19]; const float* bX2  = (const float*)d_in[20];
  const float* gX1 = (const float*)d_in[21]; const float* bnX1 = (const float*)d_in[22];
  const float* gX2 = (const float*)d_in[23]; const float* bnX2 = (const float*)d_in[24];
  const float* WE1 = (const float*)d_in[25]; const float* bE1  = (const float*)d_in[26];
  const float* WE2 = (const float*)d_in[27]; const float* bE2  = (const float*)d_in[28];
  const float* gE1 = (const float*)d_in[29]; const float* bnE1 = (const float*)d_in[30];
  const float* gE2 = (const float*)d_in[31]; const float* bnE2 = (const float*)d_in[32];

  const int RX = 4 * 256;  // 1024 node rows
  float* Xout = (float*)d_out;
  float* Eout = (float*)d_out + (size_t)RX * 256;

  // workspace bump allocator (256B aligned)
  char* ws = (char*)d_ws;
  size_t off = 0;
  auto alloc = [&](size_t bytes) -> char* {
    char* p = ws + off;
    off = (off + bytes + 255) & ~(size_t)255;
    return p;
  };
  bf16* Xbf   = (bf16*)alloc((size_t)RX * 256 * 2);
  bf16* Wq_p  = (bf16*)alloc(256 * 256 * 2);
  bf16* Wk_p  = (bf16*)alloc(256 * 256 * 2);
  bf16* Wv_p  = (bf16*)alloc(256 * 256 * 2);
  bf16* Wxo_p = (bf16*)alloc(256 * 256 * 2);
  bf16* WX1_p = (bf16*)alloc(256 * 1024 * 2);
  bf16* WX2_p = (bf16*)alloc(1024 * 256 * 2);
  bf16* Wem_p = (bf16*)alloc(64 * 256 * 2);
  bf16* Wea_p = (bf16*)alloc(64 * 256 * 2);
  bf16* Weo_p = (bf16*)alloc(256 * 64 * 2);
  bf16* WE1_p = (bf16*)alloc(64 * 128 * 2);
  bf16* WE2_p = (bf16*)alloc(128 * 64 * 2);
  float* Qf   = (float*)alloc((size_t)RX * 256 * 4);
  float* Kf   = (float*)alloc((size_t)RX * 256 * 4);
  float* Vf   = (float*)alloc((size_t)RX * 256 * 4);
  float* WVf  = (float*)alloc((size_t)RX * 256 * 4);
  bf16*  WVbf = (bf16*)alloc((size_t)RX * 256 * 2);
  float* newXf= (float*)alloc((size_t)RX * 256 * 4);
  float* X1f  = (float*)alloc((size_t)RX * 256 * 4);
  bf16*  X1bf = (bf16*)alloc((size_t)RX * 256 * 2);
  bf16*  Hxbf = (bf16*)alloc((size_t)RX * 1024 * 2);
  float* ffXf = (float*)alloc((size_t)RX * 256 * 4);

  auto cdiv = [](int a, int b) { return (a + b - 1) / b; };

  // prep: bf16 X, packed bf16 weights
  f32_to_bf16_kernel<<<cdiv(RX * 256, 256), 256, 0, stream>>>(X, Xbf, RX * 256);
  pack_b_kernel<<<cdiv(256 * 256, 256), 256, 0, stream>>>(Wq, Wq_p, 256, 256);
  pack_b_kernel<<<cdiv(256 * 256, 256), 256, 0, stream>>>(Wk, Wk_p, 256, 256);
  pack_b_kernel<<<cdiv(256 * 256, 256), 256, 0, stream>>>(Wv, Wv_p, 256, 256);
  pack_b_kernel<<<cdiv(256 * 256, 256), 256, 0, stream>>>(Wxo, Wxo_p, 256, 256);
  pack_b_kernel<<<cdiv(256 * 1024, 256), 256, 0, stream>>>(WX1, WX1_p, 256, 1024);
  pack_b_kernel<<<cdiv(1024 * 256, 256), 256, 0, stream>>>(WX2, WX2_p, 1024, 256);
  pack_b_kernel<<<cdiv(64 * 256, 256), 256, 0, stream>>>(Wem, Wem_p, 64, 256);
  pack_b_kernel<<<cdiv(64 * 256, 256), 256, 0, stream>>>(Wea, Wea_p, 64, 256);
  pack_b_kernel<<<cdiv(256 * 64, 256), 256, 0, stream>>>(Weo, Weo_p, 256, 64);
  pack_b_kernel<<<cdiv(64 * 128, 256), 256, 0, stream>>>(WE1, WE1_p, 64, 128);
  pack_b_kernel<<<cdiv(128 * 64, 256), 256, 0, stream>>>(WE2, WE2_p, 128, 64);

  // Q,K,V = (X@W + b) * node_mask  (tiles = 64*16 = 1024; 8 waves/block)
  gemm_bf16_kernel<<<128, 256, 0, stream>>>(Xbf, Wq_p, bq, nm, Qf, nullptr, RX, 256, 256, 0);
  gemm_bf16_kernel<<<128, 256, 0, stream>>>(Xbf, Wk_p, bk, nm, Kf, nullptr, RX, 256, 256, 0);
  gemm_bf16_kernel<<<128, 256, 0, stream>>>(Xbf, Wv_p, bv, nm, Vf, nullptr, RX, 256, 256, 0);

  // fused edge path: Y, newE, softmax·V, E-LN, E-FFN, Eout
  edge_kernel<<<1024, 256, 262144 + 32768 + 1024, stream>>>(
      E, nm, Qf, Kf, Vf, Wem_p, Wea_p, Weo_p, WE1_p, WE2_p,
      bem, bea, beo, bE1, bE2, gE1, bnE1, gE2, bnE2, WVf, Eout);

  // X path: newX = (WV@Wxo + b)*mask ; X1 = LN(X+newX) ; FFX ; Xout
  f32_to_bf16_kernel<<<cdiv(RX * 256, 256), 256, 0, stream>>>(WVf, WVbf, RX * 256);
  gemm_bf16_kernel<<<128, 256, 0, stream>>>(WVbf, Wxo_p, bxo, nm, newXf, nullptr, RX, 256, 256, 0);
  ln_add_kernel<<<4, 256, 0, stream>>>(X, newXf, gX1, bnX1, X1f, X1bf, RX, 256);
  gemm_bf16_kernel<<<512, 256, 0, stream>>>(X1bf, WX1_p, bX1, nullptr, nullptr, Hxbf, RX, 1024, 256, 1);
  gemm_bf16_kernel<<<128, 256, 0, stream>>>(Hxbf, WX2_p, bX2, nullptr, ffXf, nullptr, RX, 256, 1024, 0);
  ln_add_kernel<<<4, 256, 0, stream>>>(X1f, ffXf, gX2, bnX2, Xout, nullptr, RX, 256);
}
